// GatedMultiHeadSelfAttention_47991964565623
// MI455X (gfx1250) — matrix operand order repaired and checked by reading, not compiled, not run on hardware
//
#include <hip/hip_runtime.h>
#include <cstddef>

#define BATCH    2
#define SEQ      2048
#define EMB      1024
#define NHEAD    16
#define HD       64
#define GATE_EPS 1e-4f

typedef __attribute__((ext_vector_type(16))) __bf16 bf16x16;
typedef __attribute__((ext_vector_type(4)))  __bf16 bf16x4;
typedef __attribute__((ext_vector_type(8)))  float  f32x8;

// ---------- helpers ----------

__device__ inline __bf16 to_bf16(float f) { return (__bf16)f; }  // native RNE cvt

// Load 16 bf16 elements (two 16-byte chunks) into a fragment register set.
__device__ inline bf16x16 load_frag16(const __bf16* p0, const __bf16* p1) {
  union { int4 q[2]; bf16x16 v; } u;
  u.q[0] = *(const int4*)p0;
  u.q[1] = *(const int4*)p1;
  return u.v;
}

__device__ inline f32x8 wmma_bf16(bf16x16 a, bf16x16 b, f32x8 c) {
  return __builtin_amdgcn_wmma_f32_16x16x32_bf16(false, a, false, b, (short)0, c,
                                                 false, false);
}

// CDNA5 async global->LDS copy (16B per lane), tracked by ASYNCcnt.
// lds_off: wave-relative LDS byte address (= low 32 bits of generic pointer).
__device__ inline void async_copy16(unsigned lds_off, const void* gptr) {
  asm volatile("global_load_async_to_lds_b128 %0, %1, off"
               :: "v"(lds_off), "v"(gptr)
               : "memory");
}
__device__ inline void async_wait0() {
  asm volatile("s_wait_asynccnt 0x0" ::: "memory");
}
__device__ inline unsigned lds_addr(const void* p) {
  return (unsigned)(unsigned long long)p;
}

// ---------- fp32 -> bf16 conversion ----------

__global__ __launch_bounds__(256) void cvt_f32_bf16(const float* __restrict__ in,
                                                    __bf16* __restrict__ out,
                                                    int n4) {
  int i = blockIdx.x * 256 + threadIdx.x;
  if (i < n4) {
    float4 f = ((const float4*)in)[i];
    bf16x4 v;
    v[0] = to_bf16(f.x); v[1] = to_bf16(f.y);
    v[2] = to_bf16(f.z); v[3] = to_bf16(f.w);
    *(bf16x4*)&out[(size_t)i * 4] = v;
  }
}

// ---------- generic 64x64 block GEMM: C(64x64) += A(64xK) * B(Kx64), bf16 ----------
// 256 threads = 8 waves; wave (wm,wn) with wm=wave>>1 (16 rows), wn=wave&1 (32 cols).
// A tile staged via async global->LDS; B tile staged transposed [n][k] so the
// B fragment (k = 16g+i) is two contiguous ds_load_b128.
__device__ inline void gemm_block_64x64(const __bf16* __restrict__ A, int lda,
                                        const __bf16* __restrict__ B, int ldb,
                                        int K, f32x8 acc[2]) {
  __shared__ __align__(16) __bf16 As[64 * 32];   // [m][k]
  __shared__ __align__(16) __bf16 BsT[64 * 32];  // [n][k]
  const int tid  = threadIdx.x;
  const int lane = tid & 31, wave = tid >> 5;
  const int g = lane >> 4, ln = lane & 15;
  const int wm = wave >> 1, wn = wave & 1;

#pragma unroll
  for (int r = 0; r < 8; ++r) { acc[0][r] = 0.f; acc[1][r] = 0.f; }

  const int arow = tid >> 2, acol = (tid & 3) * 8;  // 64x32 tile, 16B chunks
  const int brow = tid >> 3, bcol = (tid & 7) * 8;  // 32x64 tile, 16B chunks
  const unsigned ldsA = lds_addr(&As[arow * 32 + acol]);

  for (int k0 = 0; k0 < K; k0 += 32) {
    __syncthreads();
    async_copy16(ldsA, &A[(size_t)arow * lda + k0 + acol]);
    union { int4 q; __bf16 e[8]; } bv;
    bv.q = *(const int4*)&B[(size_t)(k0 + brow) * ldb + bcol];
#pragma unroll
    for (int j = 0; j < 8; ++j) BsT[(bcol + j) * 32 + brow] = bv.e[j];
    async_wait0();
    __syncthreads();

    // A fragment: lane (g, m=ln); element i -> k = 16g+i
    const __bf16* ab = &As[(wm * 16 + ln) * 32];
    bf16x16 af = load_frag16(ab + 16 * g, ab + 16 * g + 8);

#pragma unroll
    for (int nt = 0; nt < 2; ++nt) {
      const int col = wn * 32 + nt * 16 + ln;
      const __bf16* bb = &BsT[col * 32 + 16 * g];  // element i -> k = 16g+i
      bf16x16 bfr = load_frag16(bb, bb + 8);
      acc[nt] = wmma_bf16(af, bfr, acc[nt]);
    }
  }
}

// ---------- QKV projection: per (b,h): Out = X[b] @ W[h], bf16 out ----------
__global__ __launch_bounds__(256) void gemm_qkv_kernel(const __bf16* __restrict__ X,
                                                       const __bf16* __restrict__ W,
                                                       __bf16* __restrict__ Out) {
  const int b = blockIdx.y >> 4, h = blockIdx.y & 15;
  const __bf16* A = X + (size_t)b * SEQ * EMB + (size_t)blockIdx.x * 64 * EMB;
  const __bf16* B = W + (size_t)h * EMB * HD;
  __bf16* C = Out + ((size_t)(b * NHEAD + h) * SEQ + (size_t)blockIdx.x * 64) * HD;

  f32x8 acc[2];
  gemm_block_64x64(A, EMB, B, HD, EMB, acc);

  const int lane = threadIdx.x & 31, wave = threadIdx.x >> 5;
  const int g = lane >> 4, ln = lane & 15;
  const int wm = wave >> 1, wn = wave & 1;
#pragma unroll
  for (int nt = 0; nt < 2; ++nt)
#pragma unroll
    for (int r = 0; r < 8; ++r) {
      const int row = wm * 16 + 8 * g + r;
      const int col = wn * 32 + nt * 16 + ln;
      C[(size_t)row * HD + col] = to_bf16(acc[nt][r]);
    }
}

// ---------- output projection: out = CTX @ Wo_flat, fp32 out, gate denom ----------
__global__ __launch_bounds__(256) void gemm_out_kernel(const __bf16* __restrict__ CTX,
                                                       const __bf16* __restrict__ W,
                                                       float* __restrict__ Out,
                                                       const float* __restrict__ gate) {
  const __bf16* A = CTX + (size_t)blockIdx.x * 64 * EMB;
  const __bf16* B = W + (size_t)blockIdx.y * 64;
  float* C = Out + (size_t)blockIdx.x * 64 * EMB + (size_t)blockIdx.y * 64;

  f32x8 acc[2];
  gemm_block_64x64(A, EMB, B, EMB, EMB, acc);

  float active = 0.f;
#pragma unroll
  for (int hh = 0; hh < NHEAD; ++hh) active += (gate[hh] > GATE_EPS) ? 1.f : 0.f;
  const float inv = 1.f / fmaxf(1.f, active * (1.f / NHEAD));

  const int lane = threadIdx.x & 31, wave = threadIdx.x >> 5;
  const int g = lane >> 4, ln = lane & 15;
  const int wm = wave >> 1, wn = wave & 1;
#pragma unroll
  for (int nt = 0; nt < 2; ++nt)
#pragma unroll
    for (int r = 0; r < 8; ++r) {
      const int row = wm * 16 + 8 * g + r;
      const int col = wn * 32 + nt * 16 + ln;
      C[(size_t)row * EMB + col] = acc[nt][r] * inv;
    }
}

// ---------- flash attention: per (b,h, 64-query tile); gate folded into ctx ----------
// K chunk staged row-major via async global->LDS (K^T fragment is contiguous in d);
// V chunk staged transposed [d][key] so the V B-fragment (k-dim = key) is contiguous.
__global__ __launch_bounds__(128) void attn_kernel(const __bf16* __restrict__ Q,
                                                   const __bf16* __restrict__ Kt,
                                                   const __bf16* __restrict__ V,
                                                   const float* __restrict__ mask,
                                                   const float* __restrict__ gate,
                                                   __bf16* __restrict__ ctx) {
  __shared__ __align__(16) __bf16 Ksh[32 * HD];   // [key][d]
  __shared__ __align__(16) __bf16 VshT[HD * 32];  // [d][key]
  __shared__ __align__(16) __bf16 Psh[4][16 * 32];

  const int tid  = threadIdx.x;
  const int lane = tid & 31, wave = tid >> 5;
  const int g = lane >> 4, ln = lane & 15;
  const int b = blockIdx.z, h = blockIdx.y;
  const size_t base = (size_t)(b * NHEAD + h) * SEQ * HD;
  const int q0 = blockIdx.x * 64 + wave * 16;

  // Q A-fragments for d-chunks c = 0,1 (d in [32c, 32c+32))
  bf16x16 qf[2];
#pragma unroll
  for (int c = 0; c < 2; ++c) {
    const __bf16* qp = Q + base + (size_t)(q0 + ln) * HD + 32 * c;
    qf[c] = load_frag16(qp + 16 * g, qp + 16 * g + 8);
  }

  f32x8 acc[4];
  float m_r[8], l_r[8];
#pragma unroll
  for (int r = 0; r < 8; ++r) { m_r[r] = -3.0e38f; l_r[r] = 0.f; }
#pragma unroll
  for (int nt = 0; nt < 4; ++nt)
#pragma unroll
    for (int r = 0; r < 8; ++r) acc[nt][r] = 0.f;

  const int crow = tid >> 2;         // 0..31 (key)
  const int ccol = (tid & 3) * 16;   // 0,16,32,48 (d)
  const unsigned ldsK0 = lds_addr(&Ksh[crow * HD + ccol]);
  const unsigned ldsK1 = lds_addr(&Ksh[crow * HD + ccol + 8]);

  for (int t0 = 0; t0 < SEQ; t0 += 32) {
    __syncthreads();
    {
      const __bf16* kg = Kt + base + (size_t)(t0 + crow) * HD + ccol;
      async_copy16(ldsK0, kg);
      async_copy16(ldsK1, kg + 8);

      const __bf16* vg = V + base + (size_t)(t0 + crow) * HD + ccol;
      union { int4 q; __bf16 e[8]; } v0, v1;
      v0.q = *(const int4*)vg;
      v1.q = *(const int4*)(vg + 8);
#pragma unroll
      for (int j = 0; j < 8; ++j) VshT[(ccol + j) * 32 + crow] = v0.e[j];
#pragma unroll
      for (int j = 0; j < 8; ++j) VshT[(ccol + 8 + j) * 32 + crow] = v1.e[j];
    }
    async_wait0();
    __syncthreads();

    // scores: two 16-key tiles; K^T B-fragment is contiguous in d
    f32x8 sc[2];
#pragma unroll
    for (int kt = 0; kt < 2; ++kt) {
#pragma unroll
      for (int r = 0; r < 8; ++r) sc[kt][r] = 0.f;
#pragma unroll
      for (int c = 0; c < 2; ++c) {
        const __bf16* kb = &Ksh[(kt * 16 + ln) * HD + 32 * c + 16 * g];
        bf16x16 bfr = load_frag16(kb, kb + 8);
        sc[kt] = wmma_bf16(qf[c], bfr, sc[kt]);
      }
    }

    const float mk0 = mask[(size_t)b * SEQ + t0 + ln];
    const float mk1 = mask[(size_t)b * SEQ + t0 + 16 + ln];

#pragma unroll
    for (int r = 0; r < 8; ++r) {
      float s0 = sc[0][r] * 0.125f + mk0;
      float s1 = sc[1][r] * 0.125f + mk1;
      float cm = fmaxf(s0, s1);
      cm = fmaxf(cm, __shfl_xor(cm, 1, 32));
      cm = fmaxf(cm, __shfl_xor(cm, 2, 32));
      cm = fmaxf(cm, __shfl_xor(cm, 4, 32));
      cm = fmaxf(cm, __shfl_xor(cm, 8, 32));
      float nm   = fmaxf(m_r[r], cm);
      float corr = __expf(m_r[r] - nm);
      m_r[r] = nm;
      float p0 = __expf(s0 - nm);
      float p1 = __expf(s1 - nm);
      float rs = p0 + p1;
      rs += __shfl_xor(rs, 1, 32);
      rs += __shfl_xor(rs, 2, 32);
      rs += __shfl_xor(rs, 4, 32);
      rs += __shfl_xor(rs, 8, 32);
      l_r[r] = l_r[r] * corr + rs;
#pragma unroll
      for (int nt = 0; nt < 4; ++nt) acc[nt][r] *= corr;
      // stage P into per-wave LDS (row-major 16x32) for C-layout -> A-layout
      Psh[wave][(8 * g + r) * 32 + ln]      = to_bf16(p0);
      Psh[wave][(8 * g + r) * 32 + 16 + ln] = to_bf16(p1);
    }
    asm volatile("" ::: "memory");  // keep LDS writes before re-reads (per-wave, in-order)

    const __bf16* pb = &Psh[wave][ln * 32];
    bf16x16 pf = load_frag16(pb + 16 * g, pb + 16 * g + 8);

#pragma unroll
    for (int nt = 0; nt < 4; ++nt) {
      // V B-fragment: element i -> key = 16g+i, contiguous in VshT
      const __bf16* vb = &VshT[(nt * 16 + ln) * 32 + 16 * g];
      bf16x16 vf = load_frag16(vb, vb + 8);
      acc[nt] = wmma_bf16(pf, vf, acc[nt]);
    }
  }

  const float gv = gate[h];
  const float ge = (gv >= GATE_EPS) ? gv : 0.f;
#pragma unroll
  for (int r = 0; r < 8; ++r) {
    const float inv = (l_r[r] > 0.f) ? (ge / l_r[r]) : 0.f;
    const size_t row = (size_t)b * SEQ + q0 + 8 * g + r;
#pragma unroll
    for (int nt = 0; nt < 4; ++nt)
      ctx[row * EMB + h * HD + nt * 16 + ln] = to_bf16(acc[nt][r] * inv);
  }
}

// ---------- host launch ----------

extern "C" void kernel_launch(void* const* d_in, const int* in_sizes, int n_in,
                              void* d_out, int out_size, void* d_ws, size_t ws_size,
                              hipStream_t stream) {
  const float* hidden = (const float*)d_in[0];
  const float* mask   = (const float*)d_in[1];
  const float* Wq     = (const float*)d_in[2];
  const float* Wk     = (const float*)d_in[3];
  const float* Wv     = (const float*)d_in[4];
  const float* Wo     = (const float*)d_in[5];
  const float* gate   = (const float*)d_in[6];
  float* out = (float*)d_out;

  char* w = (char*)d_ws;
  __bf16* X16  = (__bf16*)(w);                        // 8 MiB
  __bf16* Wq16 = (__bf16*)(w + ((size_t)8  << 20));   // 2 MiB
  __bf16* Wk16 = (__bf16*)(w + ((size_t)10 << 20));
  __bf16* Wv16 = (__bf16*)(w + ((size_t)12 << 20));
  __bf16* Wo16 = (__bf16*)(w + ((size_t)14 << 20));
  __bf16* Qb   = (__bf16*)(w + ((size_t)16 << 20));   // 8 MiB each
  __bf16* Kb   = (__bf16*)(w + ((size_t)24 << 20));
  __bf16* Vb   = (__bf16*)(w + ((size_t)32 << 20));
  __bf16* CTXb = (__bf16*)(w + ((size_t)40 << 20));   // 8 MiB  (total 48 MiB)

  const int nHid = BATCH * SEQ * EMB;        // 4,194,304
  const int nW   = NHEAD * EMB * HD;         // 1,048,576

  cvt_f32_bf16<<<(nHid / 4 + 255) / 256, 256, 0, stream>>>(hidden, X16, nHid / 4);
  cvt_f32_bf16<<<(nW / 4 + 255) / 256, 256, 0, stream>>>(Wq, Wq16, nW / 4);
  cvt_f32_bf16<<<(nW / 4 + 255) / 256, 256, 0, stream>>>(Wk, Wk16, nW / 4);
  cvt_f32_bf16<<<(nW / 4 + 255) / 256, 256, 0, stream>>>(Wv, Wv16, nW / 4);
  cvt_f32_bf16<<<(nW / 4 + 255) / 256, 256, 0, stream>>>(Wo, Wo16, nW / 4);

  dim3 gQKV(SEQ / 64, BATCH * NHEAD);
  gemm_qkv_kernel<<<gQKV, 256, 0, stream>>>(X16, Wq16, Qb);
  gemm_qkv_kernel<<<gQKV, 256, 0, stream>>>(X16, Wk16, Kb);
  gemm_qkv_kernel<<<gQKV, 256, 0, stream>>>(X16, Wv16, Vb);

  dim3 gAtt(SEQ / 64, NHEAD, BATCH);
  attn_kernel<<<gAtt, 128, 0, stream>>>(Qb, Kb, Vb, mask, gate, CTXb);

  dim3 gOut((BATCH * SEQ) / 64, EMB / 64);
  gemm_out_kernel<<<gOut, 256, 0, stream>>>(CTXb, Wo16, out, gate);
}
